// MinimalStreamingDurationHeadV1G_88914412961946
// MI455X (gfx1250) — matrix operand order repaired and checked by reading, not compiled, not run on hardware
//
#include <hip/hip_runtime.h>

typedef _Float16 half8  __attribute__((ext_vector_type(8)));
typedef _Float16 half16 __attribute__((ext_vector_type(16)));
typedef float    f32x8  __attribute__((ext_vector_type(8)));

#define NB   16
#define NT   2048
#define NDIM 512
#define NROWS (NB * NT)

__device__ __forceinline__ float clampf_(float x, float lo, float hi) {
    return fminf(fmaxf(x, lo), hi);
}
__device__ __forceinline__ float gelu_(float x) {
    // exact gelu (approximate=False): 0.5*x*(1+erf(x/sqrt(2)))
    return 0.5f * x * (1.0f + erff(x * 0.70710678118654752f));
}

// ---------------- K0: convert rW1[0:512,:] (row-major 512x512) to f16 -------
__global__ void k0_convert(const float* __restrict__ rW1, _Float16* __restrict__ w16) {
    int i = blockIdx.x * blockDim.x + threadIdx.x;
    if (i < 512 * 512) w16[i] = (_Float16)rW1[i];
}

// ---------------- K1: per-batch nets: spk_ctx, coarse, base_b ---------------
__global__ void k1_batch(const float* __restrict__ spk_embed, const float* __restrict__ spk_W,
                         const float* __restrict__ spk_b,   const float* __restrict__ gref,
                         const float* __restrict__ cW1,     const float* __restrict__ cb1,
                         const float* __restrict__ cW2,     const float* __restrict__ cb2,
                         const float* __restrict__ rW1,     const float* __restrict__ rb1,
                         float* __restrict__ base_b, float* __restrict__ coarse_ws) {
    __shared__ float s_vec[512];
    __shared__ float s_ctx[512];
    __shared__ float s_red[256];
    const int b = blockIdx.x, tid = threadIdx.x;

    for (int k = tid; k < 512; k += 256) s_vec[k] = spk_embed[b * 512 + k];
    __syncthreads();

    // spk_ctx = tanh(spk_embed @ spk_W + spk_b)
    for (int p = 0; p < 2; p++) {
        const int d = tid + p * 256;
        float acc = spk_b[d];
        for (int k = 0; k < 512; k++) acc += s_vec[k] * spk_W[k * 512 + d];
        s_ctx[d] = tanhf(acc);
    }
    __syncthreads();

    // coarse = 0.2*tanh(gelu([spk_ctx, g_ref] @ cW1 + cb1) @ cW2 + cb2)
    const float gr = gref[b];
    float hsum = 0.f;
    for (int p = 0; p < 2; p++) {
        const int d = tid + p * 256;
        float acc = cb1[d] + gr * cW1[512 * 512 + d];
        for (int k = 0; k < 512; k++) acc += s_ctx[k] * cW1[k * 512 + d];
        hsum += gelu_(acc) * cW2[d];
    }
    s_red[tid] = hsum;
    __syncthreads();
    for (int off = 128; off > 0; off >>= 1) {
        if (tid < off) s_red[tid] += s_red[tid + off];
        __syncthreads();
    }
    if (tid == 0) coarse_ws[b] = 0.2f * tanhf(s_red[0] + cb2[0]);

    // base_b[b,n] = spk_ctx @ rW1[512:1024,:] + rb1
    for (int p = 0; p < 2; p++) {
        const int n = tid + p * 256;
        float acc = rb1[n];
        for (int k = 0; k < 512; k++) acc += s_ctx[k] * rW1[(512 + k) * 512 + n];
        base_b[b * 512 + n] = acc;
    }
}

// ---------------- K2: EMA affine-scan + speech prefix + elementwise ---------
__global__ void k2_scan(const float* __restrict__ log_anchor, const float* __restrict__ unit_mask,
                        const float* __restrict__ sealed_mask, const float* __restrict__ silence_mask,
                        const float* __restrict__ gref, const float* __restrict__ lre,
                        const float* __restrict__ coarse_ws,
                        float* __restrict__ lr_seq, float* __restrict__ g_ws,
                        float* __restrict__ rfac_ws, float* __restrict__ speech_ws,
                        float* __restrict__ silterm_ws) {
    __shared__ float sA[256], sB[256], sS[256];
    const int b = blockIdx.x, tid = threadIdx.x;
    const int base = b * NT + tid * 8;

    float la[8], sp[8], cm[8], sc[8];
    float A = 1.f, Bc = 0.f, S = 0.f;
    #pragma unroll
    for (int i = 0; i < 8; i++) {
        const int idx = base + i;
        const float o   = log_anchor[idx];
        const float m   = clampf_(unit_mask[idx], 0.f, 1.f);
        const float se  = clampf_(sealed_mask[idx], 0.f, 1.f);
        const float sil = clampf_(silence_mask[idx], 0.f, 1.f) * m;
        const float commit = m * se;
        const float speech = commit * (1.f - sil);
        la[i] = o; sp[i] = speech; cm[i] = commit; sc[i] = commit * sil;
        // r' = a*r + b with a = 1 - c*speech, b = c*speech*o
        const float a  = 1.f - 0.05f * speech;
        const float bb = 0.05f * speech * o;
        Bc = a * Bc + bb;
        A  = a * A;
        S += speech;
    }
    sA[tid] = A; sB[tid] = Bc; sS[tid] = S;
    __syncthreads();
    // inclusive Hillis-Steele scan over affine compositions + sums
    for (int off = 1; off < 256; off <<= 1) {
        float pA = 1.f, pB = 0.f, pS = 0.f;
        const bool has = (tid >= off);
        if (has) { pA = sA[tid - off]; pB = sB[tid - off]; pS = sS[tid - off]; }
        __syncthreads();
        const float cA = sA[tid], cB = sB[tid], cS = sS[tid];
        if (has) { sA[tid] = cA * pA; sB[tid] = cA * pB + cB; sS[tid] = cS + pS; }
        __syncthreads();
    }
    const float init = lre[b];
    float r, Sexc;
    if (tid == 0) { r = init; Sexc = 0.f; }
    else          { r = sA[tid - 1] * init + sB[tid - 1]; Sexc = sS[tid - 1]; }

    const float gr = gref[b], co = coarse_ws[b];
    #pragma unroll
    for (int i = 0; i < 8; i++) {
        const int idx = base + i;
        const float lr = r;                 // seq outputs pre-update value
        lr_seq[idx] = lr;
        r += 0.05f * sp[i] * (la[i] - r);

        const float gap = clampf_(gr - lr, -0.35f, 0.35f) * cm[i];
        const float g   = (gap + co * cm[i]) * cm[i];

        const float prefix_prev = Sexc;
        Sexc += sp[i];
        const float cold = clampf_(prefix_prev * 0.5f, 0.f, 1.f);
        const float dur  = expf(la[i]);
        const float sh   = clampf_(fmaxf(dur, 1.f) - 1.f, 0.f, 1.f);
        const float tau  = 0.35f * ((dur < 2.0f) ? 0.35f : 1.0f);

        g_ws[idx]       = g;
        rfac_ws[idx]    = cold * sh * sp[i];
        speech_ws[idx]  = sp[i];
        silterm_ws[idx] = clampf_(g, -tau, tau) * sc[i];
    }
}

// ---------------- K3: query = tanh(embed[cu] + feats@fW + fb)*mask (f16) ----
__global__ void k3_query(const int* __restrict__ cu, const float* __restrict__ la_,
                         const float* __restrict__ um_, const float* __restrict__ slm_,
                         const float* __restrict__ sep_, const float* __restrict__ edge_,
                         const float* __restrict__ lr_seq, const float* __restrict__ embed,
                         const float* __restrict__ fW, const float* __restrict__ fb,
                         _Float16* __restrict__ q16) {
    const int row = blockIdx.x;
    const int tid = threadIdx.x;
    const float la  = la_[row];
    const float m   = clampf_(um_[row], 0.f, 1.f);
    const float sil = clampf_(slm_[row], 0.f, 1.f) * m;
    const float lr  = lr_seq[row];
    const float sep = sep_[row], ed = edge_[row];
    const float* e = embed + (size_t)cu[row] * 512;
    for (int d = tid; d < 512; d += 128) {
        const float s = e[d] + la * fW[d] + lr * fW[512 + d] + sil * fW[1024 + d]
                      + sep * fW[1536 + d] + ed * fW[2048 + d] + fb[d];
        q16[(size_t)row * 512 + d] = (_Float16)(tanhf(s) * m);
    }
}

// ---------------- K4: WMMA GEMM (32768x512 @ 512x512) + fused epilogue ------
// Per block: 32 M-rows. 8 waves = 2 M-subtiles x 4 N-groups of 64 columns;
// two sequential N-halves cover all 512 columns. Per wave: 4 live acc tiles
// (32 VGPRs) + explicit double-buffered A/B fragments so B loads pipeline
// ahead of the 4 back-to-back WMMAs instead of s_wait_loadcnt 0x0 per WMMA.
__global__ void __launch_bounds__(256) k4_gemm(
        const _Float16* __restrict__ q16, const _Float16* __restrict__ w16,
        const float* __restrict__ base_b, const float* __restrict__ rW1,
        const float* __restrict__ rW2, const float* __restrict__ rb2,
        const float* __restrict__ g_ws, const float* __restrict__ rfac_ws,
        const float* __restrict__ speech_ws, const float* __restrict__ silterm_ws,
        float* __restrict__ out) {
    __shared__ float sacc[32];
    const int tid  = threadIdx.x;
    const int wave = tid >> 5, lane = tid & 31;
    const int row_base = blockIdx.x * 32;         // 32 M-rows per block
    const int m_sub   = wave & 1;                 // 2 M-subtiles of 16
    const int n_group = wave >> 1;                // 4 N-groups of 64
    const int row0    = row_base + m_sub * 16;
    if (tid < 32) sacc[tid] = 0.f;
    __syncthreads();

    // A-fragment addressing (16-bit A 16x32): lane<16 -> kbase 0, lane>=16 -> kbase 8
    const int hi   = (lane >> 4) & 1;
    const int arow = row0 + (lane & 15);
    const _Float16* aptr = q16 + (size_t)arow * 512 + (hi ? 8 : 0);

    const int bb = row_base >> 11;  // batch index (tiles never straddle batches)
    const float rb2v = rb2[0];
    float gv[8];
    #pragma unroll
    for (int r = 0; r < 8; r++) gv[r] = g_ws[row0 + hi * 8 + r];

    for (int h = 0; h < 2; ++h) {
        const int n_base = h * 256 + n_group * 64;
        // B-fragment addressing (16-bit B 32x16): lane = N, element = K (table order, kbase 8*hi)
        const _Float16* bcol = w16 + (size_t)(hi * 8) * 512 + n_base + (lane & 15);

        f32x8 acc[4];
        #pragma unroll
        for (int j = 0; j < 4; j++)
            #pragma unroll
            for (int r = 0; r < 8; r++) acc[j][r] = 0.f;

        // ---- prologue: fragments for k0 = 0 ----
        half16 a_cur;
        {
            const half8 alo = *(const half8*)(aptr);
            const half8 ahi = *(const half8*)(aptr + 16);
            #pragma unroll
            for (int i = 0; i < 8; i++) { a_cur[i] = alo[i]; a_cur[i + 8] = ahi[i]; }
        }
        half16 b_cur[4];
        #pragma unroll
        for (int j = 0; j < 4; j++)
            #pragma unroll
            for (int i = 0; i < 8; i++) { b_cur[j][i] = bcol[(size_t)i * 512 + j * 16]; b_cur[j][i + 8] = bcol[(size_t)(16 + i) * 512 + j * 16]; }

        #pragma unroll
        for (int k0 = 0; k0 < 512; k0 += 32) {
            half16 a_nxt = a_cur;
            half16 b_nxt[4];
            const int k1 = k0 + 32;
            if (k1 < 512) {
                // issue next-step loads before this step's WMMAs
                const half8 alo = *(const half8*)(aptr + k1);
                const half8 ahi = *(const half8*)(aptr + k1 + 16);
                #pragma unroll
                for (int i = 0; i < 8; i++) { a_nxt[i] = alo[i]; a_nxt[i + 8] = ahi[i]; }
                const _Float16* bk = bcol + (size_t)k1 * 512;
                #pragma unroll
                for (int j = 0; j < 4; j++)
                    #pragma unroll
                    for (int i = 0; i < 8; i++) { b_nxt[j][i] = bk[(size_t)i * 512 + j * 16]; b_nxt[j][i + 8] = bk[(size_t)(16 + i) * 512 + j * 16]; }
                if (k1 + 32 < 512) __builtin_prefetch(aptr + k1 + 32, 0, 1);
            } else {
                #pragma unroll
                for (int j = 0; j < 4; j++) b_nxt[j] = b_cur[j];
            }
            #pragma unroll
            for (int j = 0; j < 4; j++) {
                acc[j] = __builtin_amdgcn_wmma_f32_16x16x32_f16(
                    false, a_cur, false, b_cur[j], (short)0, acc[j], false, false);
            }
            a_cur = a_nxt;
            #pragma unroll
            for (int j = 0; j < 4; j++) b_cur[j] = b_nxt[j];
        }

        // ---- fused epilogue for this N-half ----
        #pragma unroll
        for (int j = 0; j < 4; j++) {
            const int nIdx = n_base + j * 16 + (lane & 15);
            const float bbv = base_b[bb * 512 + nIdx];
            const float wgv = rW1[1024 * 512 + nIdx];   // rW1 last row (global_term weight)
            const float r2v = rW2[nIdx];
            #pragma unroll
            for (int r = 0; r < 8; r++) {
                const float x = acc[j][r] + bbv + gv[r] * wgv;
                float t = gelu_(x) * r2v;
                t += __shfl_xor(t, 1, 32);
                t += __shfl_xor(t, 2, 32);
                t += __shfl_xor(t, 4, 32);
                t += __shfl_xor(t, 8, 32);   // sum over the 16 N-lanes of this half
                if ((lane & 15) == 0) atomicAdd(&sacc[m_sub * 16 + hi * 8 + r], t);
            }
        }
    }
    __syncthreads();

    if (tid < 32) {
        const int row = row_base + tid;
        const float res = 0.35f * tanhf(sacc[tid] + rb2v) * rfac_ws[row];
        const float g = g_ws[row];
        out[row] = clampf_(g + res, -1.2f, 1.2f) * speech_ws[row] + silterm_ws[row];
    }
}

extern "C" void kernel_launch(void* const* d_in, const int* in_sizes, int n_in,
                              void* d_out, int out_size, void* d_ws, size_t ws_size,
                              hipStream_t stream) {
    const int*   cu        = (const int*)d_in[0];
    const float* log_anch  = (const float*)d_in[1];
    const float* unit_mask = (const float*)d_in[2];
    const float* sealed    = (const float*)d_in[3];
    const float* sep_hint  = (const float*)d_in[4];
    const float* edge_cue  = (const float*)d_in[5];
    const float* gref      = (const float*)d_in[6];
    const float* lre       = (const float*)d_in[7];
    const float* sil_mask  = (const float*)d_in[8];
    const float* spk_embed = (const float*)d_in[9];
    const float* embed     = (const float*)d_in[10];
    const float* feat_W    = (const float*)d_in[11];
    const float* feat_b    = (const float*)d_in[12];
    const float* spk_W     = (const float*)d_in[13];
    const float* spk_b     = (const float*)d_in[14];
    const float* cW1       = (const float*)d_in[15];
    const float* cb1       = (const float*)d_in[16];
    const float* cW2       = (const float*)d_in[17];
    const float* cb2       = (const float*)d_in[18];
    const float* rW1       = (const float*)d_in[19];
    const float* rb1       = (const float*)d_in[20];
    const float* rW2       = (const float*)d_in[21];
    const float* rb2       = (const float*)d_in[22];
    float* out = (float*)d_out;

    // workspace layout
    char* p = (char*)d_ws;
    _Float16* q16 = (_Float16*)p;  p += (size_t)NROWS * 512 * sizeof(_Float16);   // 33.5 MB
    _Float16* w16 = (_Float16*)p;  p += (size_t)512 * 512 * sizeof(_Float16);     // 512 KB
    float* lr_seq  = (float*)p;    p += (size_t)NROWS * sizeof(float);
    float* g_ws    = (float*)p;    p += (size_t)NROWS * sizeof(float);
    float* rfac_ws = (float*)p;    p += (size_t)NROWS * sizeof(float);
    float* speech_ws  = (float*)p; p += (size_t)NROWS * sizeof(float);
    float* silterm_ws = (float*)p; p += (size_t)NROWS * sizeof(float);
    float* base_b    = (float*)p;  p += (size_t)NB * 512 * sizeof(float);
    float* coarse_ws = (float*)p;  p += (size_t)NB * sizeof(float);

    k0_convert<<<(512 * 512 + 255) / 256, 256, 0, stream>>>(rW1, w16);
    k1_batch<<<NB, 256, 0, stream>>>(spk_embed, spk_W, spk_b, gref, cW1, cb1, cW2, cb2,
                                     rW1, rb1, base_b, coarse_ws);
    k2_scan<<<NB, 256, 0, stream>>>(log_anch, unit_mask, sealed, sil_mask, gref, lre,
                                    coarse_ws, lr_seq, g_ws, rfac_ws, speech_ws, silterm_ws);
    k3_query<<<NROWS, 128, 0, stream>>>(cu, log_anch, unit_mask, sil_mask, sep_hint, edge_cue,
                                        lr_seq, embed, feat_W, feat_b, q16);
    k4_gemm<<<NROWS / 32, 256, 0, stream>>>(q16, w16, base_b, rW1, rW2, rb2,
                                            g_ws, rfac_ws, speech_ws, silterm_ws, out);
}
